// RNNModel_8701603741907
// MI455X (gfx1250) — matrix operand order repaired and checked by reading, not compiled, not run on hardware
//
#include <hip/hip_runtime.h>
#include <hip/hip_bf16.h>
#include <math.h>

typedef __bf16 bf16;
typedef __attribute__((ext_vector_type(16))) __bf16 v16bf;
typedef __attribute__((ext_vector_type(8)))  __bf16 v8bf;
typedef __attribute__((ext_vector_type(8)))  float  v8f;

union U16 { v16bf v; v8bf h[2]; __bf16 e[16]; };

__device__ __forceinline__ v8f wmma_bf16(v16bf a, v16bf b, v8f c) {
  return __builtin_amdgcn_wmma_f32_16x16x32_bf16(false, a, false, b, (short)0, c,
                                                 false, false);
}

// ---------------- fragment loaders (contiguous-k layouts, wave32) ----------------
// A: row-major [M x K] bf16, 16x32 tile. Lane l holds row (l&15);
// lanes 0-15: K=kb+0..15; lanes 16-31: K=kb+16..31 (contiguous).
__device__ __forceinline__ v16bf load_a_frag(const bf16* __restrict__ A, int lda,
                                             int mbase, int kb, int lane) {
  int m = mbase + (lane & 15);
  int koff = (lane >> 4) << 4;                   // 0 or 16
  const bf16* p = A + (size_t)m * lda + kb + koff;
  U16 u;
  u.h[0] = *(const v8bf*)p;
  u.h[1] = *(const v8bf*)(p + 8);
  __builtin_prefetch((const void*)(p + 64), 0, 3);   // global_prefetch_b8 (near)
  return u.v;
}

// B presented as W^T where W is row-major [N x K] bf16: lane (column n) reads
// W row n contiguous in K. lanes 0-15: K=kb..kb+15; lanes 16-31: kb+16..kb+31.
__device__ __forceinline__ v16bf load_bt_frag(const bf16* __restrict__ W, int ldw,
                                              int nbase, int kb, int lane) {
  int n = nbase + (lane & 15);
  int khalf = (lane >> 4) << 4;                  // 0 or 16
  const bf16* p = W + (size_t)n * ldw + kb + khalf;
  U16 u;
  u.h[0] = *(const v8bf*)p;
  u.h[1] = *(const v8bf*)(p + 8);
  __builtin_prefetch((const void*)(p + 64), 0, 3);
  return u.v;
}

// guarded variant: lanes with n >= nmax supply zeros (tail tiles only)
__device__ __forceinline__ v16bf load_bt_frag_g(const bf16* __restrict__ W, int ldw,
                                                int nbase, int kb, int lane, int nmax) {
  int n = nbase + (lane & 15);
  int khalf = (lane >> 4) << 4;
  U16 u;
  if (n < nmax) {
    const bf16* p = W + (size_t)n * ldw + kb + khalf;
    u.h[0] = *(const v8bf*)p;
    u.h[1] = *(const v8bf*)(p + 8);
  } else {
#pragma unroll
    for (int j = 0; j < 16; ++j) u.e[j] = (__bf16)0.0f;
  }
  return u.v;
}

// ---------------- WMMA inner-loop helpers ----------------
template<bool GUARD>
__device__ __forceinline__ void mma1_loop(v8f& acc, const bf16* __restrict__ A, int lda,
                                          int mbase, const bf16* __restrict__ W, int ldw,
                                          int nbase, int Nmax, int Kd, int lane) {
  for (int kb = 0; kb < Kd; kb += 32) {
    v16bf a = load_a_frag(A, lda, mbase, kb, lane);
    v16bf b;
    if (GUARD) b = load_bt_frag_g(W, ldw, nbase, kb, lane, Nmax);
    else       b = load_bt_frag(W, ldw, nbase, kb, lane);
    acc = wmma_bf16(a, b, acc);
  }
}

// 16M x 64N per wave: one A fragment feeds 4 WMMAs (used by fused logits kernel)
template<bool GUARD>
__device__ __forceinline__ void mma4_loop(v8f (&acc)[4], const bf16* __restrict__ A,
                                          int lda, int mbase,
                                          const bf16* __restrict__ W, int ldw,
                                          int nbase, int Nmax, int Kd, int lane) {
  for (int kb = 0; kb < Kd; kb += 32) {
    v16bf a = load_a_frag(A, lda, mbase, kb, lane);
#pragma unroll
    for (int j = 0; j < 4; ++j) {
      v16bf b;
      if (GUARD) b = load_bt_frag_g(W, ldw, nbase + 16 * j, kb, lane, Nmax);
      else       b = load_bt_frag(W, ldw, nbase + 16 * j, kb, lane);
      acc[j] = wmma_bf16(a, b, acc[j]);
    }
  }
}

// 32M x 64N per wave: 8 WMMAs share 2 A-frags and 4 B-frags (1.5 loads/WMMA)
template<bool GUARD>
__device__ __forceinline__ void mma8_loop(v8f (&acc)[2][4], const bf16* __restrict__ A,
                                          int lda, int mbase,
                                          const bf16* __restrict__ W, int ldw,
                                          int nbase, int Nmax, int Kd, int lane) {
  for (int kb = 0; kb < Kd; kb += 32) {
    v16bf a0 = load_a_frag(A, lda, mbase, kb, lane);
    v16bf a1 = load_a_frag(A, lda, mbase + 16, kb, lane);
#pragma unroll
    for (int j = 0; j < 4; ++j) {
      v16bf b;
      if (GUARD) b = load_bt_frag_g(W, ldw, nbase + 16 * j, kb, lane, Nmax);
      else       b = load_bt_frag(W, ldw, nbase + 16 * j, kb, lane);
      acc[0][j] = wmma_bf16(a0, b, acc[0][j]);
      acc[1][j] = wmma_bf16(a1, b, acc[1][j]);
    }
  }
}

// ---------------- elementwise / setup kernels ----------------
__global__ void k_cvt_bf16(const float* __restrict__ src, bf16* __restrict__ dst,
                           size_t n) {
  size_t i = (size_t)blockIdx.x * blockDim.x + threadIdx.x;
  if (i < n) dst[i] = (bf16)src[i];
}

__global__ void k_gather_rows_bf16(const int* __restrict__ idx,
                                   const float* __restrict__ E,
                                   bf16* __restrict__ dst, size_t rows, int Hd) {
  size_t i = (size_t)blockIdx.x * blockDim.x + threadIdx.x;
  if (i >= rows * (size_t)Hd) return;
  size_t r = i / Hd; int h = (int)(i - r * Hd);
  dst[i] = (bf16)E[(size_t)idx[r] * Hd + h];
}

// dst[c*RP + r] = (r < R) ? bf16(src[r*C + c]) : 0   (LDS-tiled transpose+pad)
__global__ void k_transpose_pad(const float* __restrict__ src, int R, int Cc,
                                bf16* __restrict__ dst, int RP) {
  __shared__ float tile[32][33];
  int c0 = blockIdx.x * 32;
  int r0 = blockIdx.y * 32;
  int tx = threadIdx.x, ty = threadIdx.y;
  for (int i = ty; i < 32; i += 8) {
    int r = r0 + i, c = c0 + tx;
    float v = 0.f;
    if (r < R && c < Cc) v = src[(size_t)r * Cc + c];
    tile[i][tx] = v;
  }
  __syncthreads();
  for (int i = ty; i < 32; i += 8) {
    int c = c0 + i, r = r0 + tx;
    if (c < Cc && r < RP) dst[(size_t)c * RP + r] = (bf16)tile[tx][i];
  }
}

__global__ void k_wk(const float* __restrict__ mw, float* __restrict__ wk,
                     int S, int SP, int Kk) {
  int s = blockIdx.x * blockDim.x + threadIdx.x;
  if (s >= SP) return;
  if (s >= S) { for (int k = 0; k < Kk; ++k) wk[k * SP + s] = 0.f; return; }
  float m = -3.4e38f;
  for (int k = 0; k < Kk; ++k) m = fmaxf(m, mw[k * S + s]);
  float e[8]; float sum = 0.f;
  for (int k = 0; k < Kk; ++k) { e[k] = __expf(mw[k * S + s] - m); sum += e[k]; }
  float inv = 1.f / sum;
  for (int k = 0; k < Kk; ++k) wk[k * SP + s] = e[k] * inv;
}

__global__ void k_coeff(const float* __restrict__ gate, const float* __restrict__ wk,
                        bf16* __restrict__ coeffB, int Nn_, int S, int SP) {
  size_t i = (size_t)blockIdx.x * blockDim.x + threadIdx.x;
  size_t tot = (size_t)5 * Nn_ * SP;
  if (i >= tot) return;
  int sp = (int)(i % SP);
  size_t r = i / SP;
  int n = (int)(r % Nn_);
  int k = (int)(r / Nn_);
  coeffB[i] = (sp < S) ? (bf16)(gate[(size_t)n * S + sp] * wk[k * SP + sp])
                       : (bf16)0.0f;
}

__global__ void k_init_state(const float* __restrict__ h0in, const float* __restrict__ c0in,
                             float* h0f, float* h1f, float* c0f, float* c1f,
                             bf16* h0b, bf16* h1b, int BH) {
  int i = blockIdx.x * blockDim.x + threadIdx.x;
  if (i >= BH) return;
  float ha = h0in[i], hb = h0in[BH + i];
  h0f[i] = ha; h1f[i] = hb;
  h0b[i] = (bf16)ha; h1b[i] = (bf16)hb;
  c0f[i] = c0in[i]; c1f[i] = c0in[BH + i];
}

__global__ void k_final_state(const float* h0f, const float* h1f,
                              const float* c0f, const float* c1f,
                              float* outH, float* outC, int BH) {
  int i = blockIdx.x * blockDim.x + threadIdx.x;
  if (i >= BH) return;
  outH[i] = h0f[i]; outH[BH + i] = h1f[i];
  outC[i] = c0f[i]; outC[BH + i] = c1f[i];
}

__global__ void k_lstm_cell(const float* __restrict__ z,
                            float* __restrict__ c, float* __restrict__ h,
                            bf16* __restrict__ hB, bf16* __restrict__ yB,
                            int t, int Bb, int Hd) {
  int i = blockIdx.x * blockDim.x + threadIdx.x;
  if (i >= Bb * Hd) return;
  int b = i / Hd, hh = i - b * Hd;
  const float* zr = z + (size_t)b * 4 * Hd;
  float zi = zr[hh], zf = zr[Hd + hh], zg = zr[2 * Hd + hh], zo = zr[3 * Hd + hh];
  float ig = 1.f / (1.f + __expf(-zi));
  float fg = 1.f / (1.f + __expf(-zf));
  float og = 1.f / (1.f + __expf(-zo));
  float cn = fg * c[i] + ig * tanhf(zg);
  float hn = og * tanhf(cn);
  c[i] = cn; h[i] = hn;
  hB[i] = (bf16)hn;
  yB[(size_t)(t * Bb + b) * Hd + hh] = (bf16)hn;
}

// ---------------- WMMA GEMM kernels ----------------
// 1-tile/wave (max parallelism for tiny recurrent GEMM M=32)
__global__ __launch_bounds__(128)
void k_gemm_bt(const bf16* __restrict__ A, int lda,
               const bf16* __restrict__ W, int ldw,
               const float* __restrict__ bias1, const float* __restrict__ bias2,
               const float* __restrict__ addend, int ldadd,
               float* __restrict__ C, int ldc,
               int M, int N, int Kd, int act) {
  int lane = threadIdx.x & 31, wave = threadIdx.x >> 5;
  int mbase = blockIdx.y << 4;
  int nbase = (blockIdx.x << 6) + (wave << 4);
  if (mbase >= M || nbase >= N) return;
  v8f acc = {};
  if (nbase + 16 <= N)
    mma1_loop<false>(acc, A, lda, mbase, W, ldw, nbase, N, Kd, lane);
  else
    mma1_loop<true>(acc, A, lda, mbase, W, ldw, nbase, N, Kd, lane);
  int col = nbase + (lane & 15);
  if (col >= N) return;
  int rbase = mbase + ((lane >> 4) << 3);
  float badd = (bias1 ? bias1[col] : 0.f) + (bias2 ? bias2[col] : 0.f);
#pragma unroll
  for (int r = 0; r < 8; ++r) {
    int row = rbase + r;
    float v = acc[r] + badd;
    if (addend) v += addend[(size_t)row * ldadd + col];
    if (act == 1) v = 1.f / (1.f + __expf(-v));
    C[(size_t)row * ldc + col] = v;
  }
}

// 8-tile/wave (2Mx4N): block = 8 waves = 64M x 256N; for the large GEMMs
__global__ __launch_bounds__(256)
void k_gemm_bt8(const bf16* __restrict__ A, int lda,
                const bf16* __restrict__ W, int ldw,
                const float* __restrict__ bias1, const float* __restrict__ bias2,
                float* __restrict__ C, int ldc,
                int M, int N, int Kd, int act) {
  int lane = threadIdx.x & 31, wave = threadIdx.x >> 5;
  int mbase = blockIdx.y * 64 + (wave >> 2) * 32;   // wave covers mbase..mbase+31
  int nbase = blockIdx.x * 256 + (wave & 3) * 64;   // wave covers 64 columns
  if (mbase >= M || nbase >= N) return;
  v8f acc[2][4] = {{{}, {}, {}, {}}, {{}, {}, {}, {}}};
  if (nbase + 64 <= N)
    mma8_loop<false>(acc, A, lda, mbase, W, ldw, nbase, N, Kd, lane);
  else
    mma8_loop<true>(acc, A, lda, mbase, W, ldw, nbase, N, Kd, lane);
#pragma unroll
  for (int i = 0; i < 2; ++i) {
    int rbase = mbase + 16 * i + ((lane >> 4) << 3);
#pragma unroll
    for (int j = 0; j < 4; ++j) {
      int col = nbase + 16 * j + (lane & 15);
      if (col >= N) continue;
      float badd = (bias1 ? bias1[col] : 0.f) + (bias2 ? bias2[col] : 0.f);
#pragma unroll
      for (int r = 0; r < 8; ++r) {
        float v = acc[i][j][r] + badd;
        if (act == 1) v = 1.f / (1.f + __expf(-v));
        C[(size_t)(rbase + r) * ldc + col] = v;
      }
    }
  }
}

// batched over blockIdx.z, bf16 output (multi_trans path), full tiles only
__global__ __launch_bounds__(128)
void k_gemm_bt_b16(const bf16* __restrict__ A, int lda,
                   const bf16* __restrict__ Wbase, int ldw, size_t strideW,
                   bf16* __restrict__ Cbase, int ldc, size_t strideC,
                   int M, int N, int Kd) {
  const bf16* W = Wbase + (size_t)blockIdx.z * strideW;
  bf16* Cm = Cbase + (size_t)blockIdx.z * strideC;
  int lane = threadIdx.x & 31, wave = threadIdx.x >> 5;
  int mbase = blockIdx.y << 4;
  int nbase = (blockIdx.x << 6) + (wave << 4);
  if (mbase >= M || nbase >= N) return;
  v8f acc = {};
  mma1_loop<false>(acc, A, lda, mbase, W, ldw, nbase, N, Kd, lane);
  int col = nbase + (lane & 15);
  int rbase = mbase + ((lane >> 4) << 3);
#pragma unroll
  for (int r = 0; r < 8; ++r)
    Cm[(size_t)(rbase + r) * ldc + col] = (bf16)acc[r];
}

// fused: logits[n,z] = sum_k (mt_k.ctx)[n,z]*(coeff_k.M1)[n,z] + sbias[z]
__global__ __launch_bounds__(256)
void k_logits(const bf16* __restrict__ mtB,    // [5,Nn,512]
              const bf16* __restrict__ ctxB,   // [Z,512]  (W layout)
              const bf16* __restrict__ coeffB, // [5,Nn,2208]
              const bf16* __restrict__ M1T,    // [Z,2208] (W layout)
              const float* __restrict__ sbias, // [Z]
              float* __restrict__ logits, int Nn_, int Zd) {
  int lane = threadIdx.x & 31, wave = threadIdx.x >> 5;
  int mbase = blockIdx.y * 32 + (wave >> 2) * 16;
  int zbase = blockIdx.x * 256 + (wave & 3) * 64;
  if (zbase >= Zd) return;
  bool full = (zbase + 64 <= Zd);
  v8f acc[4] = {{}, {}, {}, {}};
  for (int k = 0; k < 5; ++k) {
    const bf16* Amt = mtB + (size_t)k * Nn_ * 512;
    const bf16* Aco = coeffB + (size_t)k * Nn_ * 2208;
    v8f P[4] = {{}, {}, {}, {}};
    v8f Q[4] = {{}, {}, {}, {}};
    if (full) {
      mma4_loop<false>(P, Amt, 512, mbase, ctxB, 512, zbase, Zd, 512, lane);
      mma4_loop<false>(Q, Aco, 2208, mbase, M1T, 2208, zbase, Zd, 2208, lane);
    } else {
      mma4_loop<true>(P, Amt, 512, mbase, ctxB, 512, zbase, Zd, 512, lane);
      mma4_loop<true>(Q, Aco, 2208, mbase, M1T, 2208, zbase, Zd, 2208, lane);
    }
#pragma unroll
    for (int j = 0; j < 4; ++j) acc[j] += P[j] * Q[j];
  }
  int rbase = mbase + ((lane >> 4) << 3);
#pragma unroll
  for (int j = 0; j < 4; ++j) {
    int zt = zbase + 16 * j;
    if (zt >= Zd) continue;
    int col = zt + (lane & 15);
    float sb = sbias[col];
#pragma unroll
    for (int r = 0; r < 8; ++r)
      logits[(size_t)(rbase + r) * Zd + col] = acc[j][r] + sb;
  }
}

__global__ void k_softmax_row(const float* __restrict__ logits,
                              bf16* __restrict__ senseB, int Zd) {
  __shared__ float red[256];
  int row = blockIdx.x, tid = threadIdx.x;
  const float* p = logits + (size_t)row * Zd;
  float m = -3.4e38f;
  for (int z = tid; z < Zd; z += 256) m = fmaxf(m, p[z]);
  red[tid] = m; __syncthreads();
  for (int s = 128; s > 0; s >>= 1) {
    if (tid < s) red[tid] = fmaxf(red[tid], red[tid + s]);
    __syncthreads();
  }
  m = red[0]; __syncthreads();
  float sum = 0.f;
  for (int z = tid; z < Zd; z += 256) sum += __expf(p[z] - m);
  red[tid] = sum; __syncthreads();
  for (int s = 128; s > 0; s >>= 1) {
    if (tid < s) red[tid] += red[tid + s];
    __syncthreads();
  }
  float inv = 1.f / red[0];
  bf16* q = senseB + (size_t)row * Zd;
  for (int z = tid; z < Zd; z += 256) q[z] = (bf16)(__expf(p[z] - m) * inv);
}

// ---------------- host launcher ----------------
static inline int cdiv(int a, int b) { return (a + b - 1) / b; }

extern "C" void kernel_launch(void* const* d_in, const int* in_sizes, int n_in,
                              void* d_out, int out_size, void* d_ws, size_t ws_size,
                              hipStream_t stream) {
  (void)in_sizes; (void)n_in; (void)out_size; (void)ws_size;
  constexpr int T = 32, Bb = 32, H = 512, V = 10000, S = 2186, Z = 20000, Kk = 5;
  constexpr int Nn = T * Bb, H4 = 4 * H, SP = 2208, BH = Bb * H;

  const int*   ids   = (const int*)  d_in[0];
  const float* h0in  = (const float*)d_in[1];
  const float* c0in  = (const float*)d_in[2];
  const float* E     = (const float*)d_in[3];
  const float* Wih0  = (const float*)d_in[4];
  const float* Whh0  = (const float*)d_in[5];
  const float* bih0  = (const float*)d_in[6];
  const float* bhh0  = (const float*)d_in[7];
  const float* Wih1  = (const float*)d_in[8];
  const float* Whh1  = (const float*)d_in[9];
  const float* bih1  = (const float*)d_in[10];
  const float* bhh1  = (const float*)d_in[11];
  const float* Wg    = (const float*)d_in[12];
  const float* bg    = (const float*)d_in[13];
  const float* MT    = (const float*)d_in[14];
  const float* mw    = (const float*)d_in[15];
  const float* sbias = (const float*)d_in[16];
  const int*   widx  = (const int*)  d_in[17];
  const float* M1    = (const float*)d_in[18];
  const float* M2    = (const float*)d_in[19];

  // workspace bump allocator (~690 MB; M2T dominates)
  char* base = (char*)d_ws;
  size_t off = 0;
  auto alloc = [&](size_t bytes) -> void* {
    void* p = base + off;
    off = (off + bytes + 255) & ~(size_t)255;
    return p;
  };
  bf16*  embB   = (bf16*) alloc((size_t)Nn * H * 2);
  bf16*  Wih0B  = (bf16*) alloc((size_t)H4 * H * 2);
  bf16*  Whh0B  = (bf16*) alloc((size_t)H4 * H * 2);
  bf16*  Wih1B  = (bf16*) alloc((size_t)H4 * H * 2);
  bf16*  Whh1B  = (bf16*) alloc((size_t)H4 * H * 2);
  bf16*  WgB    = (bf16*) alloc((size_t)S * H * 2);
  bf16*  MTT    = (bf16*) alloc((size_t)Kk * H * H * 2);   // [k][h,d] = MT[k][d,h]
  bf16*  ctxB   = (bf16*) alloc((size_t)Z * H * 2);        // [z,h] = E[widx[z],h]
  bf16*  M1T    = (bf16*) alloc((size_t)Z * SP * 2);       // [z,sp] = M1[sp,z] (pad)
  bf16*  M2T    = (bf16*) alloc((size_t)V * Z * 2);        // [v,z]  = M2[z,v]
  float* Xpre0  = (float*)alloc((size_t)Nn * H4 * 4);
  float* Xpre1  = (float*)alloc((size_t)Nn * H4 * 4);
  float* zbuf   = (float*)alloc((size_t)Bb * H4 * 4);
  float* h0f    = (float*)alloc((size_t)BH * 4);
  float* c0f    = (float*)alloc((size_t)BH * 4);
  float* h1f    = (float*)alloc((size_t)BH * 4);
  float* c1f    = (float*)alloc((size_t)BH * 4);
  bf16*  h0b    = (bf16*) alloc((size_t)BH * 2);
  bf16*  h1b    = (bf16*) alloc((size_t)BH * 2);
  bf16*  y0B    = (bf16*) alloc((size_t)Nn * H * 2);
  bf16*  outB   = (bf16*) alloc((size_t)Nn * H * 2);
  bf16*  mtB    = (bf16*) alloc((size_t)Kk * Nn * H * 2);
  float* wkbuf  = (float*)alloc((size_t)Kk * SP * 4);
  bf16*  coeffB = (bf16*) alloc((size_t)Kk * Nn * SP * 2);
  float* logitsW= (float*)alloc((size_t)Nn * Z * 4);
  bf16*  senseB = (bf16*) alloc((size_t)Nn * Z * 2);

  float* out_word = (float*)d_out;                 // [Nn,V]
  float* out_gate = out_word + (size_t)Nn * V;     // [Nn,S]
  float* out_h    = out_gate + (size_t)Nn * S;     // [2,Bb,H]
  float* out_c    = out_h    + (size_t)2 * BH;     // [2,Bb,H]

  dim3 tb(32, 8);

  // --- conversions / gathers / transposes ---
  k_cvt_bf16<<<cdiv(H4 * H, 256), 256, 0, stream>>>(Wih0, Wih0B, (size_t)H4 * H);
  k_cvt_bf16<<<cdiv(H4 * H, 256), 256, 0, stream>>>(Whh0, Whh0B, (size_t)H4 * H);
  k_cvt_bf16<<<cdiv(H4 * H, 256), 256, 0, stream>>>(Wih1, Wih1B, (size_t)H4 * H);
  k_cvt_bf16<<<cdiv(H4 * H, 256), 256, 0, stream>>>(Whh1, Whh1B, (size_t)H4 * H);
  k_cvt_bf16<<<cdiv(S * H, 256), 256, 0, stream>>>(Wg, WgB, (size_t)S * H);
  k_gather_rows_bf16<<<cdiv(Nn * H, 256), 256, 0, stream>>>(ids, E, embB, Nn, H);
  k_gather_rows_bf16<<<(int)(((size_t)Z * H + 255) / 256), 256, 0, stream>>>(widx, E, ctxB, Z, H);
  for (int k = 0; k < Kk; ++k)
    k_transpose_pad<<<dim3(16, 16), tb, 0, stream>>>(MT + (size_t)k * H * H, H, H,
                                                     MTT + (size_t)k * H * H, H);
  k_transpose_pad<<<dim3(cdiv(Z, 32), cdiv(SP, 32)), tb, 0, stream>>>(M1, S, Z, M1T, SP);
  k_transpose_pad<<<dim3(cdiv(V, 32), cdiv(Z, 32)), tb, 0, stream>>>(M2, Z, V, M2T, Z);
  k_wk<<<cdiv(SP, 256), 256, 0, stream>>>(mw, wkbuf, S, SP, Kk);
  k_init_state<<<cdiv(BH, 256), 256, 0, stream>>>(h0in, c0in, h0f, h1f, c0f, c1f,
                                                  h0b, h1b, BH);

  // --- LSTM layer 0 ---
  k_gemm_bt8<<<dim3(cdiv(H4, 256), Nn / 64), 256, 0, stream>>>(
      embB, H, Wih0B, H, bih0, bhh0, Xpre0, H4, Nn, H4, H, 0);
  for (int t = 0; t < T; ++t) {
    k_gemm_bt<<<dim3(cdiv(H4, 64), Bb / 16), 128, 0, stream>>>(
        h0b, H, Whh0B, H, nullptr, nullptr, Xpre0 + (size_t)t * Bb * H4, H4,
        zbuf, H4, Bb, H4, H, 0);
    k_lstm_cell<<<cdiv(BH, 256), 256, 0, stream>>>(zbuf, c0f, h0f, h0b, y0B, t, Bb, H);
  }

  // --- LSTM layer 1 ---
  k_gemm_bt8<<<dim3(cdiv(H4, 256), Nn / 64), 256, 0, stream>>>(
      y0B, H, Wih1B, H, bih1, bhh1, Xpre1, H4, Nn, H4, H, 0);
  for (int t = 0; t < T; ++t) {
    k_gemm_bt<<<dim3(cdiv(H4, 64), Bb / 16), 128, 0, stream>>>(
        h1b, H, Whh1B, H, nullptr, nullptr, Xpre1 + (size_t)t * Bb * H4, H4,
        zbuf, H4, Bb, H4, H, 0);
    k_lstm_cell<<<cdiv(BH, 256), 256, 0, stream>>>(zbuf, c1f, h1f, h1b, outB, t, Bb, H);
  }
  k_final_state<<<cdiv(BH, 256), 256, 0, stream>>>(h0f, h1f, c0f, c1f, out_h, out_c, BH);

  // --- gate = sigmoid(out @ Wg^T + bg) -> d_out ---
  k_gemm_bt8<<<dim3(cdiv(S, 256), Nn / 64), 256, 0, stream>>>(
      outB, H, WgB, H, bg, nullptr, out_gate, S, Nn, S, H, 1);

  // --- coeff_k = bf16(gate * softmax_k(multi_weight)), padded to SP ---
  k_coeff<<<(int)(((size_t)Kk * Nn * SP + 255) / 256), 256, 0, stream>>>(
      out_gate, wkbuf, coeffB, Nn, S, SP);

  // --- mt_k = out @ multi_trans[k] (B pre-transposed -> vector loads) ---
  k_gemm_bt_b16<<<dim3(H / 64, Nn / 16, Kk), 128, 0, stream>>>(
      outB, H, MTT, H, (size_t)H * H, mtB, H, (size_t)Nn * H, Nn, H, H);

  // --- fused logits = sum_k (mt_k.ctx)*(coeff_k.M1) + sense_bias ---
  k_logits<<<dim3(cdiv(Z, 256), Nn / 32), 256, 0, stream>>>(
      mtB, ctxB, coeffB, M1T, sbias, logitsW, Nn, Z);

  // --- softmax over Z -> bf16 probs ---
  k_softmax_row<<<Nn, 256, 0, stream>>>(logitsW, senseB, Z);

  // --- word_prob = sense_prob @ M2 (pre-transposed bf16) -> d_out ---
  k_gemm_bt8<<<dim3(cdiv(V, 256), Nn / 64), 256, 0, stream>>>(
      senseB, Z, M2T, Z, nullptr, nullptr, out_word, V, Nn, V, Z, 0);
}
